// GraphAttentionLayer_38405597561527
// MI455X (gfx1250) — matrix operand order repaired and checked by reading, not compiled, not run on hardware
//
#include <hip/hip_runtime.h>
#include <hip/hip_fp16.h>

typedef __attribute__((ext_vector_type(16))) _Float16 v16h;
typedef __attribute__((ext_vector_type(8)))  _Float16 v8h;
typedef __attribute__((ext_vector_type(4)))  _Float16 v4h;
typedef __attribute__((ext_vector_type(8)))  float    v8f;
typedef __attribute__((ext_vector_type(4)))  float    v4f;

#define NN     8192
#define DD     256
#define ALPHA  0.2f
#define LPAD   136   // padded LDS row stride in halves (272B: conflict-free b128 frags)

// ds_swizzle xor-add (imm must be ICE at call site -> macro)
#define SWZ_ADD(v, imm) \
  (v) += __int_as_float(__builtin_amdgcn_ds_swizzle(__float_as_int(v), (imm)))

__device__ __forceinline__ v16h cat8(v8h lo, v8h hi) {
  return __builtin_shufflevector(lo, hi, 0,1,2,3,4,5,6,7,8,9,10,11,12,13,14,15);
}

// workgroup-relative LDS byte offset (ptrtoint of addrspace(3) pointer)
__device__ __forceinline__ unsigned lds_off_u32(const void* p) {
  return (unsigned)(unsigned long long)(const __attribute__((address_space(3))) char*)p;
}

// async global->LDS 16B copy (ASYNCcnt-tracked, no VGPR transit)
__device__ __forceinline__ void async_ld_b128(unsigned lds_byte_off, const void* gptr) {
  asm volatile("global_load_async_to_lds_b128 %0, %1, off"
               :: "v"(lds_byte_off), "v"(gptr) : "memory");
}
__device__ __forceinline__ void wait_async0() {
  asm volatile("s_wait_asynccnt 0x0" ::: "memory");
}

// ---------------- kernel 0: W [k][n] f32 -> WT [n][k] f16 ----------------
__global__ void k_prep_wt(const float* __restrict__ W, _Float16* __restrict__ WT) {
  int n = threadIdx.x;
  int k0 = blockIdx.x * 4;
  for (int r = 0; r < 4; ++r) {
    int k = k0 + r;
    WT[n * DD + k] = (_Float16)W[k * DD + n];
  }
}

// ---------------- kernel 1: Wh = h@W via WMMA; emit WhT(f16), Wh1, Wh2 ----
__launch_bounds__(256)
__global__ void k_gemm1(const float* __restrict__ h, const _Float16* __restrict__ WT,
                        _Float16* __restrict__ WhT, const float* __restrict__ a,
                        float* __restrict__ Wh1, float* __restrict__ Wh2) {
  __shared__ _Float16 wt_lds[DD][LPAD];          // 69,632 B

  const int tid  = threadIdx.x;
  const int lane = tid & 31;
  const int w    = tid >> 5;                     // 8 waves
  const int m0   = blockIdx.x * 128 + w * 16;    // 16 rows per wave
  const int r    = lane & 15;
  const int khA  = (lane < 16) ? 0 : 8;          // A-matrix K-half select
  const int khB  = (lane < 16) ? 0 : 8;          // B-matrix K-half select

  v8f acc[16];
  #pragma unroll
  for (int nt = 0; nt < 16; ++nt) { v8f z = {}; acc[nt] = z; }

  for (int kc = 0; kc < DD; kc += 128) {
    __syncthreads();
    // stage WT[:, kc:kc+128] -> LDS via async copies
    for (int it = 0; it < 16; ++it) {
      int l = it * 256 + tid;
      int n = l >> 4;
      int u = l & 15;
      async_ld_b128(lds_off_u32(&wt_lds[n][u * 8]), WT + n * DD + kc + u * 8);
    }
    wait_async0();
    __syncthreads();

    #pragma unroll
    for (int sub = 0; sub < 4; ++sub) {
      // A fragment: row (m0+r), K = kc+sub*32+khA + {0..7, 16..23}, f32->f16
      const float* hp = h + (m0 + r) * DD + kc + sub * 32 + khA;
      v4f a0 = *(const v4f*)(hp);
      v4f a1 = *(const v4f*)(hp + 4);
      v4f a2 = *(const v4f*)(hp + 16);
      v4f a3 = *(const v4f*)(hp + 20);
      v16h afrag;
      #pragma unroll
      for (int i = 0; i < 4; ++i) {
        afrag[i]      = (_Float16)a0[i];
        afrag[4 + i]  = (_Float16)a1[i];
        afrag[8 + i]  = (_Float16)a2[i];
        afrag[12 + i] = (_Float16)a3[i];
      }
      #pragma unroll
      for (int nt = 0; nt < 16; ++nt) {
        const _Float16* bp = &wt_lds[nt * 16 + r][sub * 32 + khB];
        v16h bfrag = cat8(*(const v8h*)bp, *(const v8h*)(bp + 16));
        acc[nt] = __builtin_amdgcn_wmma_f32_16x16x32_f16(
            false, afrag, false, bfrag, (short)0, acc[nt], false, false);
      }
    }
  }

  // epilogue: WhT (transposed f16) + Wh1/Wh2 projections
  #pragma unroll
  for (int nt = 0; nt < 16; ++nt) {
    int n = nt * 16 + r;
    #pragma unroll
    for (int v = 0; v < 8; ++v) {
      int row = m0 + v + ((lane < 16) ? 0 : 8);
      WhT[n * NN + row] = (_Float16)acc[nt][v];
    }
  }
  float a1v[16], a2v[16];
  #pragma unroll
  for (int nt = 0; nt < 16; ++nt) {
    a1v[nt] = a[nt * 16 + r];
    a2v[nt] = a[DD + nt * 16 + r];
  }
  #pragma unroll
  for (int v = 0; v < 8; ++v) {
    float p1 = 0.f, p2 = 0.f;
    #pragma unroll
    for (int nt = 0; nt < 16; ++nt) {
      p1 += acc[nt][v] * a1v[nt];
      p2 += acc[nt][v] * a2v[nt];
    }
    // reduce across the 16-lane half (rows live in lanes 0-15 / 16-31)
    SWZ_ADD(p1, 0x041F); SWZ_ADD(p1, 0x081F); SWZ_ADD(p1, 0x101F); SWZ_ADD(p1, 0x201F);
    SWZ_ADD(p2, 0x041F); SWZ_ADD(p2, 0x081F); SWZ_ADD(p2, 0x101F); SWZ_ADD(p2, 0x201F);
    if (r == 0) {
      int row = m0 + v + ((lane < 16) ? 0 : 8);
      Wh1[row] = p1;
      Wh2[row] = p2;
    }
  }
}

// ---------------- kernel 2: global max of Wh2 -----------------------------
__global__ void k_max(const float* __restrict__ Wh2, float* __restrict__ m2) {
  __shared__ float red[256];
  float m = -1e30f;
  for (int i = threadIdx.x; i < NN; i += 256) m = fmaxf(m, Wh2[i]);
  red[threadIdx.x] = m;
  __syncthreads();
  for (int s = 128; s > 0; s >>= 1) {
    if (threadIdx.x < s) red[threadIdx.x] = fmaxf(red[threadIdx.x], red[threadIdx.x + s]);
    __syncthreads();
  }
  if (threadIdx.x == 0) *m2 = red[0];
}

// ---------------- kernel 3: fused masked-softmax attention @ Wh + ELU -----
__launch_bounds__(256)
__global__ void k_attn(const int* __restrict__ adj, const _Float16* __restrict__ WhT,
                       const float* __restrict__ Wh1, const float* __restrict__ Wh2,
                       const float* __restrict__ m2p, float* __restrict__ out) {
  __shared__ _Float16 wt_lds[2][DD][LPAD];  // double-buffered WhT tile: 2x69,632 B
  __shared__ _Float16 p_lds[64][LPAD];      // prob tile (row x k): 17,408 B
  __shared__ float    sums_lds[64];

  const int tid  = threadIdx.x;
  const int lane = tid & 31;
  const int w    = tid >> 5;                 // 8 waves
  const int row0 = blockIdx.x * 64;          // 64 query rows per WG
  const int r    = lane & 15;
  // phase-2 tiling: wave -> (16-row group, 128-col group)
  const int wrow0 = (w >> 1) * 16;
  const int n0    = (w & 1) * 128;
  const int khA   = (lane < 16) ? 0 : 8;
  const int khB   = (lane < 16) ? 0 : 8;
  // this thread's staging slice (fixed across chunks)
  const int sn = tid >> 4;                   // rows tid/16, tid/16+16, ... (16 units/row)
  const int su = tid & 15;

  const float m2 = *m2p;
  float wh1r[8], mi[8], rowsum[8];
  #pragma unroll
  for (int rr = 0; rr < 8; ++rr) {          // phase-1 rows for this wave
    float x = Wh1[row0 + w * 8 + rr];
    wh1r[rr] = x;
    float t = x + m2;                       // upper bound on unmasked scores
    mi[rr] = (t > 0.f) ? t : ALPHA * t;
    rowsum[rr] = 0.f;
  }
  v8f acc[8];
  #pragma unroll
  for (int nt = 0; nt < 8; ++nt) { v8f z = {}; acc[nt] = z; }

  // prologue: async-stage chunk 0 into buffer 0
  for (int it = 0; it < 16; ++it) {
    int n = it * 16 + sn;
    async_ld_b128(lds_off_u32(&wt_lds[0][n][su * 8]), WhT + n * NN + su * 8);
  }

  for (int kc = 0; kc < NN; kc += 128) {
    const int b = (kc >> 7) & 1;
    // phase 1: masked probabilities for rows w*8..w*8+7 (coalesced adj reads)
    v4f wh2v = *(const v4f*)(Wh2 + kc + lane * 4);
    #pragma unroll
    for (int rr = 0; rr < 8; ++rr) {
      int row = row0 + w * 8 + rr;
      const int* ap = adj + (long)row * NN + kc + lane * 4;
      int4 av = *(const int4*)ap;
      if (kc + 128 < NN) __builtin_prefetch(ap + 128, 0, 0);  // next adj chunk
      const int* am = (const int*)&av;
      v4h pv;
      float s = 0.f;
      #pragma unroll
      for (int e = 0; e < 4; ++e) {
        float x = wh1r[rr] + wh2v[e];
        x = (x > 0.f) ? x : ALPHA * x;       // LeakyReLU
        float p = am[e] ? __expf(x - mi[rr]) : 0.f;
        s += p;
        pv[e] = (_Float16)p;
      }
      rowsum[rr] += s;
      *(v4h*)&p_lds[w * 8 + rr][lane * 4] = pv;
    }
    wait_async0();                           // staging for THIS chunk complete
    __syncthreads();                         // p_lds + wt_lds[b] visible to all

    // async-stage NEXT chunk into the other buffer (overlaps phase 2 + next phase 1);
    // safe: wt_lds[b^1] readers (phase 2 of kc-128) all passed the barrier above.
    if (kc + 128 < NN) {
      for (int it = 0; it < 16; ++it) {
        int n = it * 16 + sn;
        async_ld_b128(lds_off_u32(&wt_lds[b ^ 1][n][su * 8]),
                      WhT + n * NN + (kc + 128) + su * 8);
      }
    }

    // phase 2: P(16x32 f16) x WhT(32x16 f16) WMMAs from LDS fragments
    #pragma unroll
    for (int sub = 0; sub < 4; ++sub) {
      const _Float16* ap = &p_lds[wrow0 + r][sub * 32 + khA];
      v16h afrag = cat8(*(const v8h*)ap, *(const v8h*)(ap + 16));
      #pragma unroll
      for (int nt = 0; nt < 8; ++nt) {
        const _Float16* bp = &wt_lds[b][n0 + nt * 16 + r][sub * 32 + khB];
        v16h bfrag = cat8(*(const v8h*)bp, *(const v8h*)(bp + 16));
        acc[nt] = __builtin_amdgcn_wmma_f32_16x16x32_f16(
            false, afrag, false, bfrag, (short)0, acc[nt], false, false);
      }
    }
    __syncthreads();                         // phase-2 reads done before next p_lds writes
  }

  // row sums: full 32-lane reduction, publish via LDS
  #pragma unroll
  for (int rr = 0; rr < 8; ++rr) {
    float s = rowsum[rr];
    SWZ_ADD(s, 0x041F); SWZ_ADD(s, 0x081F); SWZ_ADD(s, 0x101F);
    SWZ_ADD(s, 0x201F); SWZ_ADD(s, 0x401F);
    if (lane == 0) sums_lds[w * 8 + rr] = s;
  }
  __syncthreads();

  float inv[8];
  #pragma unroll
  for (int v = 0; v < 8; ++v) {
    int rl = wrow0 + v + ((lane < 16) ? 0 : 8);
    inv[v] = 1.0f / sums_lds[rl];
  }
  #pragma unroll
  for (int nt = 0; nt < 8; ++nt) {
    #pragma unroll
    for (int v = 0; v < 8; ++v) {
      int rl = wrow0 + v + ((lane < 16) ? 0 : 8);
      float x = acc[nt][v] * inv[v];
      x = (x > 0.f) ? x : expm1f(x);         // ELU
      out[(long)(row0 + rl) * DD + n0 + nt * 16 + r] = x;
    }
  }
}

// ---------------- host launch --------------------------------------------
extern "C" void kernel_launch(void* const* d_in, const int* in_sizes, int n_in,
                              void* d_out, int out_size, void* d_ws, size_t ws_size,
                              hipStream_t stream) {
  const float* h   = (const float*)d_in[0];   // [8192,256]
  const int*   adj = (const int*)d_in[1];     // [8192,8192]
  const float* W   = (const float*)d_in[2];   // [256,256]
  const float* a   = (const float*)d_in[3];   // [512,1]
  float* out = (float*)d_out;                 // [8192,256]

  char* ws = (char*)d_ws;
  _Float16* WhT = (_Float16*)ws;                               // 4 MB: [256][8192] f16
  _Float16* WT  = (_Float16*)(ws + 4u * 1024 * 1024);          // 128 KB: [256][256] f16
  float* Wh1 = (float*)(ws + 4u * 1024 * 1024 + 128u * 1024);  // 32 KB
  float* Wh2 = Wh1 + NN;                                       // 32 KB
  float* m2  = Wh2 + NN;                                       // 4 B

  k_prep_wt<<<64, 256, 0, stream>>>(W, WT);
  k_gemm1  <<<64, 256, 0, stream>>>(h, WT, WhT, a, Wh1, Wh2);
  k_max    <<<1, 256, 0, stream>>>(Wh2, m2);
  k_attn   <<<128, 256, 0, stream>>>(adj, WhT, Wh1, Wh2, m2, out);
}
